// MultiHeadAttentionLowLevel_44521630990378
// MI455X (gfx1250) — matrix operand order repaired and checked by reading, not compiled, not run on hardware
//
#include <hip/hip_runtime.h>
#include <hip/hip_bf16.h>

// ---------------------------------------------------------------------------
// MultiHeadAttention for MI455X (gfx1250)
//   B=4, S=2048, D=1024, H=16, DK=64
//   1) qkv_proj (x3): fp32 -> bf16 via v_cvt_pk_bf16_f32, v_wmma_f32_16x16x32_bf16
//   2) flash-attention: K/V tiles staged in LDS via GLOBAL_LOAD_ASYNC_TO_LDS_B128
//      (double-buffered, ASYNCcnt), P transposed C->A layout through LDS
//   3) out_proj: bf16 @ bf16(W_o) -> fp32 d_out
// ---------------------------------------------------------------------------

typedef __bf16 bf16_t;
typedef __attribute__((ext_vector_type(16))) __bf16 v16bf;
typedef __attribute__((ext_vector_type(8)))  __bf16 v8bf;
typedef __attribute__((ext_vector_type(8)))  float  v8f;
typedef __attribute__((ext_vector_type(4)))  float  v4f;

// exact parameter type of __builtin_amdgcn_global_load_async_to_lds_b128
typedef int v4i_vs __attribute__((vector_size(4 * sizeof(int))));
typedef __attribute__((address_space(1))) v4i_vs* gp128_t;   // global (AS1)
typedef __attribute__((address_space(3))) v4i_vs* lp128_t;   // LDS (AS3)

#define WMMA_BF16(a, b, c) \
    __builtin_amdgcn_wmma_f32_16x16x32_bf16(false, (a), false, (b), (short)0, (c), false, false)

#if __has_builtin(__builtin_amdgcn_global_load_async_to_lds_b128)
#define ASYNC_LDS 1
#endif

constexpr int Bb = 4;
constexpr int Ss = 2048;
constexpr int Dd = 1024;
constexpr int Hh = 16;
constexpr int DKk = 64;

// 16-byte global -> LDS copy: async DMA (ASYNCcnt) when available, else direct.
__device__ __forceinline__ void cp16_g2l(const bf16_t* g, bf16_t* l)
{
#ifdef ASYNC_LDS
    __builtin_amdgcn_global_load_async_to_lds_b128((gp128_t)g, (lp128_t)l, 0, 0);
#else
    *(v8bf*)l = *(const v8bf*)g;
#endif
}

__device__ __forceinline__ void wait_async_all()
{
#ifdef ASYNC_LDS
#if __has_builtin(__builtin_amdgcn_s_wait_asynccnt)
    __builtin_amdgcn_s_wait_asynccnt(0);
#else
    asm volatile("s_wait_asynccnt 0x0" ::: "memory");
#endif
#endif
}

// ---------------------------------------------------------------------------
// Kernel 1: Y = X @ W^T  (X: [8192,1024] f32, W: [1024,1024] f32, row-major)
// which: 0 = Q (scale 0.125, [B,H,S,DK]) 1 = K ([B,H,S,DK]) 2 = V^T ([B,H,DK,S])
// Wave tile 16x64; block = 8 waves -> 128x64.
// ---------------------------------------------------------------------------
__global__ __launch_bounds__(256) void qkv_proj_kernel(
    const float* __restrict__ X, const float* __restrict__ W,
    bf16_t* __restrict__ out, int which)
{
    const int lane = threadIdx.x & 31;
    const int wv   = threadIdx.x >> 5;
    const int hi   = lane >> 4;
    const int ln   = lane & 15;
    const int m0   = blockIdx.y * 128 + wv * 16;
    const int n0   = blockIdx.x * 64;

    v8f acc[4] = {};

    for (int k0 = 0; k0 < Dd; k0 += 32) {
        v16bf a;  // A frag: row = ln, K runs [kb,kb+8) & [kb+16,kb+24), kb = 8*hi
        {
            const float* pa = X + (size_t)(m0 + ln) * Dd + k0 + hi * 8;
            v4f f0 = *(const v4f*)(pa);
            v4f f1 = *(const v4f*)(pa + 4);
            v4f f2 = *(const v4f*)(pa + 16);
            v4f f3 = *(const v4f*)(pa + 20);
#pragma unroll
            for (int i = 0; i < 4; ++i) {
                a[i]      = (bf16_t)f0[i];
                a[4 + i]  = (bf16_t)f1[i];
                a[8 + i]  = (bf16_t)f2[i];
                a[12 + i] = (bf16_t)f3[i];
            }
        }
#pragma unroll
        for (int nt = 0; nt < 4; ++nt) {   // B frag: col = ln, K runs [kb,kb+8) & [kb+16,kb+24), kb = 8*hi
            const float* pb = W + (size_t)(n0 + nt * 16 + ln) * Dd + k0 + hi * 8;
            v16bf b;
#pragma unroll
            for (int i = 0; i < 16; i += 4) {
                v4f f = *(const v4f*)(pb + i + (i & 8));
                b[i]     = (bf16_t)f[0];
                b[i + 1] = (bf16_t)f[1];
                b[i + 2] = (bf16_t)f[2];
                b[i + 3] = (bf16_t)f[3];
            }
            acc[nt] = WMMA_BF16(a, b, acc[nt]);
        }
    }

    const float qscale = (which == 0) ? 0.125f : 1.0f;  // fold 1/sqrt(DK) into Q
#pragma unroll
    for (int nt = 0; nt < 4; ++nt) {
        const int n  = n0 + nt * 16 + ln;
        const int h  = n >> 6;
        const int dk = n & 63;
#pragma unroll
        for (int r = 0; r < 8; ++r) {
            const int m = m0 + r + 8 * hi;   // C layout: M = r + 8*hi, N = ln
            const int b = m >> 11;
            const int s = m & 2047;
            const float v = acc[nt][r] * qscale;
            size_t idx;
            if (which == 2) idx = ((size_t)(b * Hh + h) * DKk + dk) * Ss + s;   // V^T
            else            idx = ((size_t)(b * Hh + h) * Ss + s) * DKk + dk;   // Q/K
            out[idx] = (bf16_t)v;
        }
    }
}

// ---------------------------------------------------------------------------
// Kernel 2: flash attention. Grid (S/128, B*H), 8 waves x 16 query rows.
// K/V 64-key tiles async-DMA'd to LDS (double buffered), shared by all waves.
// ---------------------------------------------------------------------------
__global__ __launch_bounds__(256) void attention_kernel(
    const bf16_t* __restrict__ Qb, const bf16_t* __restrict__ Kb,
    const bf16_t* __restrict__ Vt, const unsigned char* __restrict__ mask,
    bf16_t* __restrict__ AO)
{
    // rows padded to 72 elems (144B = 9x16B): async chunks stay 16B-aligned,
    // fragment reads land on 16 distinct banks (36r mod 64).
    __shared__ bf16_t kt[2][64][72];    // K tile:  row = key, col = dk
    __shared__ bf16_t vt[2][64][72];    // V^T tile: row = dk, col = key-in-block
    __shared__ bf16_t plds[8][16][72];  // per-wave P tile (C->A layout transpose)

    const int tid  = threadIdx.x;
    const int lane = tid & 31;
    const int wv   = tid >> 5;
    const int hi   = lane >> 4;
    const int ln   = lane & 15;
    const int bh   = blockIdx.y;
    const int b    = bh >> 4;
    const int h    = bh & 15;
    const int q0   = blockIdx.x * 128 + wv * 16;

    const bf16_t* Qh = Qb + (size_t)bh * Ss * DKk;
    const bf16_t* Kh = Kb + (size_t)bh * Ss * DKk;
    const bf16_t* Vh = Vt + (size_t)bh * DKk * Ss;

    // staging slice owned by this thread: 2 rows per tile, 16B chunk each
    const int chunk = tid & 7;          // 8 x 16B chunks per 64-elem row
    const int row0  = tid >> 3;         // rows 0..31
    const int row1  = row0 + 32;        // rows 32..63

    // ---- preload Q A-fragments (Q pre-scaled by 1/8)
    v16bf qf[2];
#pragma unroll
    for (int ks = 0; ks < 2; ++ks) {
        const bf16_t* pq = Qh + (size_t)(q0 + ln) * DKk + ks * 32 + hi * 8;
        v8bf l0 = *(const v8bf*)(pq);
        v8bf l1 = *(const v8bf*)(pq + 16);
#pragma unroll
        for (int i = 0; i < 8; ++i) { qf[ks][i] = l0[i]; qf[ks][8 + i] = l1[i]; }
    }

    v8f o[4] = {};
    float mrow[8], lrow[8];
#pragma unroll
    for (int r = 0; r < 8; ++r) { mrow[r] = -__builtin_inff(); lrow[r] = 0.0f; }

    const float LOG2E = 1.4426950408889634f;
    constexpr int NBLK = Ss / 64;  // 32 key blocks

    // ---- prologue: stage key block 0 into buffer 0
    {
        const int kb = 0;
        cp16_g2l(Kh + (size_t)(kb + row0) * DKk + chunk * 8, &kt[0][row0][chunk * 8]);
        cp16_g2l(Kh + (size_t)(kb + row1) * DKk + chunk * 8, &kt[0][row1][chunk * 8]);
        cp16_g2l(Vh + (size_t)row0 * Ss + kb + chunk * 8,    &vt[0][row0][chunk * 8]);
        cp16_g2l(Vh + (size_t)row1 * Ss + kb + chunk * 8,    &vt[0][row1][chunk * 8]);
    }

    for (int j = 0; j < NBLK; ++j) {
        const int kb  = j * 64;
        const int cur = j & 1;

        wait_async_all();    // block-j DMA (this wave's share) complete
        __syncthreads();     // whole tile visible; everyone done with buffer (j+1)&1

        if (j + 1 < NBLK) {  // overlap block j+1 DMA with block j compute
            const int kbn = kb + 64, nb = (j + 1) & 1;
            cp16_g2l(Kh + (size_t)(kbn + row0) * DKk + chunk * 8, &kt[nb][row0][chunk * 8]);
            cp16_g2l(Kh + (size_t)(kbn + row1) * DKk + chunk * 8, &kt[nb][row1][chunk * 8]);
            cp16_g2l(Vh + (size_t)row0 * Ss + kbn + chunk * 8,    &vt[nb][row0][chunk * 8]);
            cp16_g2l(Vh + (size_t)row1 * Ss + kbn + chunk * 8,    &vt[nb][row1][chunk * 8]);
        }

        // ---- scores: S_tile[16x64] = Q(16x64) . K^T(64x64), K frags from LDS
        v8f sc[4] = {};
#pragma unroll
        for (int nt = 0; nt < 4; ++nt) {
#pragma unroll
            for (int ks = 0; ks < 2; ++ks) {
                const bf16_t* pk = &kt[cur][nt * 16 + ln][ks * 32 + hi * 8];
                v8bf k0 = *(const v8bf*)(pk);
                v8bf k1 = *(const v8bf*)(pk + 16);
                v16bf kf;
#pragma unroll
                for (int i = 0; i < 8; ++i) { kf[i] = k0[i]; kf[8 + i] = k1[i]; }
                sc[nt] = WMMA_BF16(qf[ks], kf, sc[nt]);
            }
        }

        // ---- key mask (column-wise; harness mask is all-false)
#pragma unroll
        for (int nt = 0; nt < 4; ++nt) {
            if (mask[(size_t)b * Ss + kb + nt * 16 + ln]) {
#pragma unroll
                for (int r = 0; r < 8; ++r) sc[nt][r] = -__builtin_inff();
            }
        }

        // ---- online softmax (row q = r + 8*hi lives on this half's 16 lanes)
        float alpha[8];
#pragma unroll
        for (int r = 0; r < 8; ++r) {
            float mx = fmaxf(fmaxf(sc[0][r], sc[1][r]), fmaxf(sc[2][r], sc[3][r]));
#pragma unroll
            for (int d = 1; d < 16; d <<= 1) mx = fmaxf(mx, __shfl_xor(mx, d, 32));
            const float mnew = fmaxf(mrow[r], mx);
            alpha[r] = exp2f((mrow[r] - mnew) * LOG2E);
            mrow[r]  = mnew;
        }
#pragma unroll
        for (int r = 0; r < 8; ++r) {
            float sum = 0.0f;
#pragma unroll
            for (int nt = 0; nt < 4; ++nt) {
                const float p = exp2f((sc[nt][r] - mrow[r]) * LOG2E);
                sc[nt][r] = p;
                sum += p;
            }
#pragma unroll
            for (int d = 1; d < 16; d <<= 1) sum += __shfl_xor(sum, d, 32);
            lrow[r] = lrow[r] * alpha[r] + sum;
        }
#pragma unroll
        for (int nt = 0; nt < 4; ++nt)
#pragma unroll
            for (int r = 0; r < 8; ++r) o[nt][r] *= alpha[r];

        // ---- P: C-layout -> LDS -> A-layout (bf16); wave-private region
#pragma unroll
        for (int nt = 0; nt < 4; ++nt)
#pragma unroll
            for (int r = 0; r < 8; ++r)
                plds[wv][r + 8 * hi][nt * 16 + ln] = (bf16_t)sc[nt][r];

        asm volatile("s_wait_dscnt 0x0" ::: "memory");  // wave-internal LDS RAW

        v16bf pf[2];
#pragma unroll
        for (int ks = 0; ks < 2; ++ks) {
            const bf16_t* pp = &plds[wv][ln][ks * 32 + hi * 8];
            v8bf l0 = *(const v8bf*)(pp);
            v8bf l1 = *(const v8bf*)(pp + 16);
#pragma unroll
            for (int i = 0; i < 8; ++i) { pf[ks][i] = l0[i]; pf[ks][8 + i] = l1[i]; }
        }

        // ---- O += P(16x64) . V(64x64), V frags from LDS (row = dk, col = key)
#pragma unroll
        for (int nt = 0; nt < 4; ++nt) {
#pragma unroll
            for (int ks = 0; ks < 2; ++ks) {
                const bf16_t* pv = &vt[cur][nt * 16 + ln][ks * 32 + hi * 8];
                v8bf v0 = *(const v8bf*)(pv);
                v8bf v1 = *(const v8bf*)(pv + 16);
                v16bf vf;
#pragma unroll
                for (int i = 0; i < 8; ++i) { vf[i] = v0[i]; vf[8 + i] = v1[i]; }
                o[nt] = WMMA_BF16(pf[ks], vf, o[nt]);
            }
        }
    }

    // ---- normalize, emit bf16 attn-out [B,S,D]
#pragma unroll
    for (int nt = 0; nt < 4; ++nt) {
#pragma unroll
        for (int r = 0; r < 8; ++r) {
            const int q  = q0 + r + 8 * hi;
            const int dk = nt * 16 + ln;
            const float val = o[nt][r] / lrow[r];
            AO[((size_t)(b * Ss + q)) * Dd + h * DKk + dk] = (bf16_t)val;
        }
    }
}

// ---------------------------------------------------------------------------
// Kernel 3: d_out = AO(bf16) @ W_o^T, fp32 result.
// ---------------------------------------------------------------------------
__global__ __launch_bounds__(256) void out_proj_kernel(
    const bf16_t* __restrict__ AO, const float* __restrict__ Wo,
    float* __restrict__ Y)
{
    const int lane = threadIdx.x & 31;
    const int wv   = threadIdx.x >> 5;
    const int hi   = lane >> 4;
    const int ln   = lane & 15;
    const int m0   = blockIdx.y * 128 + wv * 16;
    const int n0   = blockIdx.x * 64;

    v8f acc[4] = {};

    for (int k0 = 0; k0 < Dd; k0 += 32) {
        v16bf a;
        {
            const bf16_t* pa = AO + (size_t)(m0 + ln) * Dd + k0 + hi * 8;
            v8bf l0 = *(const v8bf*)(pa);
            v8bf l1 = *(const v8bf*)(pa + 16);
#pragma unroll
            for (int i = 0; i < 8; ++i) { a[i] = l0[i]; a[8 + i] = l1[i]; }
        }
#pragma unroll
        for (int nt = 0; nt < 4; ++nt) {
            const float* pb = Wo + (size_t)(n0 + nt * 16 + ln) * Dd + k0 + hi * 8;
            v16bf b;
#pragma unroll
            for (int i = 0; i < 16; i += 4) {
                v4f f = *(const v4f*)(pb + i + (i & 8));
                b[i]     = (bf16_t)f[0];
                b[i + 1] = (bf16_t)f[1];
                b[i + 2] = (bf16_t)f[2];
                b[i + 3] = (bf16_t)f[3];
            }
            acc[nt] = WMMA_BF16(a, b, acc[nt]);
        }
    }

#pragma unroll
    for (int nt = 0; nt < 4; ++nt)
#pragma unroll
        for (int r = 0; r < 8; ++r)
            Y[(size_t)(m0 + r + 8 * hi) * Dd + n0 + nt * 16 + ln] = acc[nt][r];
}

// ---------------------------------------------------------------------------
extern "C" void kernel_launch(void* const* d_in, const int* in_sizes, int n_in,
                              void* d_out, int out_size, void* d_ws, size_t ws_size,
                              hipStream_t stream)
{
    (void)in_sizes; (void)n_in; (void)out_size; (void)ws_size;

    const float* query = (const float*)d_in[0];
    const float* key   = (const float*)d_in[1];
    const float* value = (const float*)d_in[2];
    const unsigned char* mask = (const unsigned char*)d_in[3];
    const float* Wq = (const float*)d_in[4];
    const float* Wk = (const float*)d_in[5];
    const float* Wv = (const float*)d_in[6];
    const float* Wo = (const float*)d_in[7];

    const size_t headElems = (size_t)Bb * Hh * Ss * DKk;   // 8,388,608
    bf16_t* qbf = (bf16_t*)d_ws;
    bf16_t* kbf = qbf + headElems;
    bf16_t* vtb = kbf + headElems;
    bf16_t* aob = vtb + headElems;                         // 64 MB workspace

    const dim3 blk(256);
    const dim3 gProj(Dd / 64, (Bb * Ss) / 128);            // 16 x 64
    const dim3 gAttn(Ss / 128, Bb * Hh);                   // 16 x 64

    qkv_proj_kernel<<<gProj, blk, 0, stream>>>(query, Wq, qbf, 0);
    qkv_proj_kernel<<<gProj, blk, 0, stream>>>(key,   Wk, kbf, 1);
    qkv_proj_kernel<<<gProj, blk, 0, stream>>>(value, Wv, vtb, 2);
    attention_kernel<<<gAttn, blk, 0, stream>>>(qbf, kbf, vtb, mask, aob);
    out_proj_kernel<<<gProj, blk, 0, stream>>>(aob, Wo, (float*)d_out);
}
